// ChimeraBlock_31937376813256
// MI455X (gfx1250) — compile-verified, matrix operand order checked by reading, not run on hardware
//
#include <hip/hip_runtime.h>
#include <hip/hip_bf16.h>
#include <math.h>

// ---------------------------------------------------------------------------
// ChimeraBlock (RG-LRU + SwiGLU) for MI455X / gfx1250
// Compute-bound (~421 GFLOP, AI > 2000 FLOP/B) -> all GEMMs via
// v_wmma_f32_16x16x32_bf16 (fp32 accumulate). Wave tile 32x64 (8 accum),
// block 8 waves = 256x64. bf16 weights (~103 MB) stay resident in the
// 192 MB L2, so direct global_load_b128 fragment loads are the right path.
// ---------------------------------------------------------------------------

typedef __attribute__((ext_vector_type(16))) __bf16 v16bf;
typedef __attribute__((ext_vector_type(8)))  __bf16 v8bf;
typedef __attribute__((ext_vector_type(8)))  float  v8f;

#define DD   2048
#define HH   2048
#define FFND 5632
#define BB   2
#define TT   2048
#define MM   (BB * TT)   // 4096 rows

// ------------------------- helpers -----------------------------------------

__device__ __forceinline__ float sigf(float x) {
    return 1.0f / (1.0f + __expf(-x));
}

__device__ __forceinline__ v16bf load_frag(const __bf16* __restrict__ p) {
    // Lane fragment: K[0..7] and K[16..23] relative to p (the +8 half-select
    // for lanes 16..31 is folded into p by the caller). Two b128 loads.
    v8bf lo = *reinterpret_cast<const v8bf*>(p);
    v8bf hi = *reinterpret_cast<const v8bf*>(p + 16);
    v16bf r;
#pragma unroll
    for (int i = 0; i < 8; ++i) { r[i] = lo[i]; r[i + 8] = hi[i]; }
    return r;
}

// ------------------------- fp32 -> bf16 convert -----------------------------

__global__ __launch_bounds__(256)
void cvt_f32_bf16(const float* __restrict__ src, __bf16* __restrict__ dst,
                  size_t n) {
    size_t i = (size_t)blockIdx.x * blockDim.x + threadIdx.x;
    size_t stride = (size_t)gridDim.x * blockDim.x;
    for (; i < n; i += stride) dst[i] = (__bf16)src[i];
}

// ------------------------- RMSNorm (fp32 in -> bf16 out) --------------------

__global__ __launch_bounds__(256)
void rmsnorm_bf16(const float* __restrict__ x, const float* __restrict__ w,
                  __bf16* __restrict__ out, int cols) {
    const int row = blockIdx.x;
    const float* xr = x + (size_t)row * cols;
    float s = 0.0f;
    for (int c = threadIdx.x; c < cols; c += 256) {
        float v = xr[c];
        s += v * v;
    }
    __shared__ float red[256];
    red[threadIdx.x] = s;
    __syncthreads();
#pragma unroll
    for (int off = 128; off > 0; off >>= 1) {
        if (threadIdx.x < off) red[threadIdx.x] += red[threadIdx.x + off];
        __syncthreads();
    }
    const float rinv = rsqrtf(red[0] / (float)cols + 1e-6f);
    __bf16* orow = out + (size_t)row * cols;
    for (int c = threadIdx.x; c < cols; c += 256)
        orow[c] = (__bf16)(xr[c] * rinv * w[c]);
}

// ------------------------- WMMA GEMM ----------------------------------------
// out[m,n] = sum_k A[m,k] * W[n,k]   (A: [M,K] bf16, W: [N,K] bf16, out fp32)
// Epilogues: 0 none | 1 sigmoid(v + bias[n]) | 2 silu(v) | 3 v + res[m*N+n]
// Wave tile: 32 x 64 (2 A-frags x 4 B-frags -> 8 accumulators).
// Block: 8 waves -> 256 x 64 output tile.

template <int EPI>
__global__ __launch_bounds__(256)
void gemm_bf16_wmma(const __bf16* __restrict__ A, const __bf16* __restrict__ W,
                    float* __restrict__ out, const float* __restrict__ aux,
                    int M, int N, int K) {
    const int lane  = threadIdx.x & 31;
    const int wave  = threadIdx.x >> 5;
    const int half  = (lane >> 4) & 1;        // K-half select per lane group
    const int l15   = lane & 15;
    const int tileN = blockIdx.x * 64;
    const int m0    = blockIdx.y * 256 + wave * 32;

    const __bf16* A0 = A + (size_t)(m0 + l15) * K + half * 8;
    const __bf16* A1 = A0 + (size_t)16 * K;
    const __bf16* B0 = W + (size_t)(tileN + l15) * K + half * 8;
    const __bf16* B1 = B0 + (size_t)16 * K;
    const __bf16* B2 = B0 + (size_t)32 * K;
    const __bf16* B3 = B0 + (size_t)48 * K;

    v8f acc[2][4];
#pragma unroll
    for (int r = 0; r < 2; ++r)
#pragma unroll
        for (int j = 0; j < 4; ++j) acc[r][j] = (v8f){};

    for (int k = 0; k < K; k += 32) {
        v16bf a0 = load_frag(A0 + k);
        v16bf a1 = load_frag(A1 + k);
        v16bf b0 = load_frag(B0 + k);
        v16bf b1 = load_frag(B1 + k);
        v16bf b2 = load_frag(B2 + k);
        v16bf b3 = load_frag(B3 + k);
        acc[0][0] = __builtin_amdgcn_wmma_f32_16x16x32_bf16(false, a0, false, b0,
                                                            (short)0, acc[0][0], false, false);
        acc[1][0] = __builtin_amdgcn_wmma_f32_16x16x32_bf16(false, a1, false, b0,
                                                            (short)0, acc[1][0], false, false);
        acc[0][1] = __builtin_amdgcn_wmma_f32_16x16x32_bf16(false, a0, false, b1,
                                                            (short)0, acc[0][1], false, false);
        acc[1][1] = __builtin_amdgcn_wmma_f32_16x16x32_bf16(false, a1, false, b1,
                                                            (short)0, acc[1][1], false, false);
        acc[0][2] = __builtin_amdgcn_wmma_f32_16x16x32_bf16(false, a0, false, b2,
                                                            (short)0, acc[0][2], false, false);
        acc[1][2] = __builtin_amdgcn_wmma_f32_16x16x32_bf16(false, a1, false, b2,
                                                            (short)0, acc[1][2], false, false);
        acc[0][3] = __builtin_amdgcn_wmma_f32_16x16x32_bf16(false, a0, false, b3,
                                                            (short)0, acc[0][3], false, false);
        acc[1][3] = __builtin_amdgcn_wmma_f32_16x16x32_bf16(false, a1, false, b3,
                                                            (short)0, acc[1][3], false, false);
    }

    // C/D layout: VGPR i -> row (m0 + r*16 + half*8 + i),
    //             col = tileN + j*16 + (lane&15)
#pragma unroll
    for (int r = 0; r < 2; ++r) {
#pragma unroll
        for (int j = 0; j < 4; ++j) {
            const int n = tileN + j * 16 + l15;
#pragma unroll
            for (int i = 0; i < 8; ++i) {
                const int m = m0 + r * 16 + half * 8 + i;
                const size_t o = (size_t)m * N + n;
                float v = acc[r][j][i];
                if (EPI == 1) v = sigf(v + aux[n]);
                else if (EPI == 2) v = v * sigf(v);
                else if (EPI == 3) v = v + aux[o];
                out[o] = v;
            }
        }
    }
}

// ------------------------- RG-LRU gate combine ------------------------------
// in-place: rg (r -> a), xp (x_proj -> gated); reads ig (already sigmoided).

__global__ __launch_bounds__(256)
void gate_combine(const float* __restrict__ ig, float* __restrict__ rg,
                  float* __restrict__ xp, const float* __restrict__ lam,
                  size_t total) {
    size_t i = (size_t)blockIdx.x * blockDim.x + threadIdx.x;
    size_t stride = (size_t)gridDim.x * blockDim.x;
    for (; i < total; i += stride) {
        const int h = (int)(i % HH);
        const float log_a = __logf(sigf(lam[h]) + 1e-8f);
        const float a = __expf(8.0f * rg[i] * log_a);
        const float g = sqrtf(1.0f - a * a + 1e-6f) * (ig[i] * xp[i]);
        rg[i] = a;
        xp[i] = g;
    }
}

// ------------------------- linear recurrence scan ---------------------------

__global__ __launch_bounds__(256)
void lru_scan(const float* __restrict__ a, const float* __restrict__ g,
              const float* __restrict__ h0, float* __restrict__ hs) {
    const int tid = blockIdx.x * blockDim.x + threadIdx.x;
    if (tid >= BB * HH) return;
    const int b = tid / HH;
    const int h = tid % HH;
    float s = h0[h];
    for (int t = 0; t < TT; ++t) {
        const size_t idx = ((size_t)(b * TT + t)) * HH + h;
        s = fmaf(a[idx], s, g[idx]);
        hs[idx] = s;
    }
}

// ------------------------- g*u -> bf16 --------------------------------------

__global__ __launch_bounds__(256)
void mul_to_bf16(const float* __restrict__ g, const float* __restrict__ u,
                 __bf16* __restrict__ p, size_t n) {
    size_t i = (size_t)blockIdx.x * blockDim.x + threadIdx.x;
    size_t stride = (size_t)gridDim.x * blockDim.x;
    for (; i < n; i += stride) p[i] = (__bf16)(g[i] * u[i]);
}

// ---------------------------------------------------------------------------

extern "C" void kernel_launch(void* const* d_in, const int* in_sizes, int n_in,
                              void* d_out, int out_size, void* d_ws, size_t ws_size,
                              hipStream_t stream) {
    const float* x         = (const float*)d_in[0];
    const float* norm1_w   = (const float*)d_in[1];
    const float* rec_in_w  = (const float*)d_in[2];
    const float* rec_ig_w  = (const float*)d_in[3];
    const float* rec_ig_b  = (const float*)d_in[4];
    const float* rec_rg_w  = (const float*)d_in[5];
    const float* rec_rg_b  = (const float*)d_in[6];
    const float* rec_lam   = (const float*)d_in[7];
    const float* rec_out_w = (const float*)d_in[8];
    const float* rec_h0    = (const float*)d_in[9];
    const float* rec_nw    = (const float*)d_in[10];
    const float* norm2_w   = (const float*)d_in[11];
    const float* ffn_gw    = (const float*)d_in[12];
    const float* ffn_uw    = (const float*)d_in[13];
    const float* ffn_dw    = (const float*)d_in[14];
    float* out = (float*)d_out;

    // ---------------- workspace carve-up (256B aligned) ----------------
    char* ws = (char*)d_ws;
    size_t off = 0;
    auto carve = [&](size_t bytes) -> char* {
        char* p = ws + off;
        off += (bytes + 255) & ~(size_t)255;
        return p;
    };
    const size_t nHD = (size_t)HH * DD;      // 4.19M
    const size_t nFD = (size_t)FFND * DD;    // 11.5M
    const size_t nMH = (size_t)MM * HH;      // 8.39M
    const size_t nMF = (size_t)MM * FFND;    // 23.1M

    __bf16* w_in_bf   = (__bf16*)carve(nHD * 2);
    __bf16* w_ig_bf   = (__bf16*)carve(nHD * 2);
    __bf16* w_rg_bf   = (__bf16*)carve(nHD * 2);
    __bf16* w_out_bf  = (__bf16*)carve(nHD * 2);
    __bf16* w_gate_bf = (__bf16*)carve(nFD * 2);
    __bf16* w_up_bf   = (__bf16*)carve(nFD * 2);
    __bf16* w_down_bf = (__bf16*)carve(nFD * 2);
    __bf16* hbuf      = (__bf16*)carve(nMH * 2); // h_in -> hs_norm -> h2
    float*  xpbuf     = (float*)carve(nMH * 4);  // x_proj -> gated -> x_mid
    float*  igbuf     = (float*)carve(nMH * 4);  // i_t -> hs
    float*  rgbuf     = (float*)carve(nMH * 4);  // r_t -> a_t
    float*  gbuf      = (float*)carve(nMF * 4);  // silu(gate)
    float*  ubuf      = (float*)carve(nMF * 4);  // up
    __bf16* pbuf      = (__bf16*)carve(nMF * 2); // g*u bf16
    (void)ws_size; (void)n_in; (void)in_sizes; (void)out_size;

    // ---------------- weight conversion (one pass, ~13us of BW) -------------
    auto cvt = [&](const float* s, __bf16* d, size_t n) {
        cvt_f32_bf16<<<4096, 256, 0, stream>>>(s, d, n);
    };
    cvt(rec_in_w,  w_in_bf,   nHD);
    cvt(rec_ig_w,  w_ig_bf,   nHD);
    cvt(rec_rg_w,  w_rg_bf,   nHD);
    cvt(rec_out_w, w_out_bf,  nHD);
    cvt(ffn_gw,    w_gate_bf, nFD);
    cvt(ffn_uw,    w_up_bf,   nFD);
    cvt(ffn_dw,    w_down_bf, nFD);

    const dim3 blk(256);
    const dim3 gH(HH / 64, MM / 256);     // N=2048 GEMMs
    const dim3 gF(FFND / 64, MM / 256);   // N=5632 GEMMs

    // 1) h_in = rmsnorm(x, norm1_w)  [bf16]
    rmsnorm_bf16<<<MM, blk, 0, stream>>>(x, norm1_w, hbuf, DD);

    // 2) x_proj / i_t / r_t GEMMs (sigmoid+bias fused for gates)
    gemm_bf16_wmma<0><<<gH, blk, 0, stream>>>(hbuf, w_in_bf, xpbuf, nullptr,
                                              MM, HH, DD);
    gemm_bf16_wmma<1><<<gH, blk, 0, stream>>>(hbuf, w_ig_bf, igbuf, rec_ig_b,
                                              MM, HH, DD);
    gemm_bf16_wmma<1><<<gH, blk, 0, stream>>>(hbuf, w_rg_bf, rgbuf, rec_rg_b,
                                              MM, HH, DD);

    // 3) a_t / gated (in place over rgbuf / xpbuf)
    gate_combine<<<16384, blk, 0, stream>>>(igbuf, rgbuf, xpbuf, rec_lam,
                                            nMH);

    // 4) sequential RG-LRU scan -> hs (igbuf)
    lru_scan<<<(BB * HH + 255) / 256, blk, 0, stream>>>(rgbuf, xpbuf, rec_h0,
                                                        igbuf);

    // 5) hs_norm = rmsnorm(hs, rec_norm_w) [bf16, reuse hbuf]
    rmsnorm_bf16<<<MM, blk, 0, stream>>>(igbuf, rec_nw, hbuf, HH);

    // 6) x_mid = x + hs_norm @ rec_out_w^T  (residual fused; xpbuf free now)
    gemm_bf16_wmma<3><<<gH, blk, 0, stream>>>(hbuf, w_out_bf, xpbuf, x,
                                              MM, DD, HH);

    // 7) h2 = rmsnorm(x_mid, norm2_w) [bf16, reuse hbuf]
    rmsnorm_bf16<<<MM, blk, 0, stream>>>(xpbuf, norm2_w, hbuf, DD);

    // 8) gate (silu fused) and up GEMMs
    gemm_bf16_wmma<2><<<gF, blk, 0, stream>>>(hbuf, w_gate_bf, gbuf, nullptr,
                                              MM, FFND, DD);
    gemm_bf16_wmma<0><<<gF, blk, 0, stream>>>(hbuf, w_up_bf, ubuf, nullptr,
                                              MM, FFND, DD);

    // 9) p = silu(g) * u  [bf16]
    mul_to_bf16<<<32768, blk, 0, stream>>>(gbuf, ubuf, pbuf, nMF);

    // 10) out = x_mid + p @ ffn_down_w^T (residual fused)
    gemm_bf16_wmma<3><<<gH, blk, 0, stream>>>(pbuf, w_down_bf, out, xpbuf,
                                              MM, DD, FFND);
}
